// LSTMModel_77988016161295
// MI455X (gfx1250) — compile-verified, matrix operand order checked by reading, not run on hardware
//
#include <hip/hip_runtime.h>
#include <hip/hip_fp16.h>

// Problem dims
#define B_    64
#define S_    1024
#define F_    128
#define H_    512
#define V_    64
#define T_    64
#define GATES 2048          // 4*H
#define NWG   32            // persistent workgroups; each owns 16 hidden units
#define BLK   256           // 8 wave32 per WG

// LDS layout (dynamic shared)
#define SM_WHH   0          // 64KB  f16 W_hh fragments (enc, then dec)
#define SM_AUX   65536      // 64KB  f16 W_ih fragments (enc) / out_W fragments (WG0, dec)
#define SM_XLOG  131072     // 16KB  f16 x_t buffer (enc) / f32 logits (dec)
#define SM_GBUF  147456     // 16KB  f32 gates [4][64][16]
#define SM_CBUF  163840     // 4KB   f32 cell state [64][16]
#define SM_BIAS  167936     // 256B  f32 combined bias [64]
#define SMEM_BYTES 168192

typedef __attribute__((ext_vector_type(16))) _Float16 v16h;
typedef __attribute__((ext_vector_type(8)))  _Float16 v8h;
typedef __attribute__((ext_vector_type(8)))  float    v8f;
typedef __attribute__((ext_vector_type(4)))  unsigned u32x4;
typedef __attribute__((ext_vector_type(8)))  unsigned u32x8;

static __device__ __forceinline__ v8f wmma_f16(v16h a, v16h b, v8f c) {
  return __builtin_amdgcn_wmma_f32_16x16x32_f16(false, a, false, b, (short)0, c,
                                                false, false);
}

// A-fragment (16x32 f16) for 16-bit WMMA: lane l<16 -> row M=l&15, K in
// {0..7,16..23}; lane>=16 -> K in {8..15,24..31}. Two contiguous 16B loads.
static __device__ __forceinline__ v16h load_a_frag(const _Float16* base, int ld,
                                                   int m_tile, int kt, int lane) {
  int row = m_tile * 16 + (lane & 15);
  int k0  = kt * 32 + ((lane >> 4) << 3);
  const _Float16* p = base + row * ld + k0;
  v8h lo = *(const v8h*)(p);
  v8h hi = *(const v8h*)(p + 16);
  return __builtin_shufflevector(lo, hi, 0,1,2,3,4,5,6,7,8,9,10,11,12,13,14,15);
}

// B-fragment from LDS, pre-packed in fragment-linear order (32B per lane).
static __device__ __forceinline__ v16h load_b_frag(const _Float16* frags, int kt,
                                                   int ntl, int lane) {
  return *(const v16h*)(frags + ((((kt << 2) + ntl) * 32 + lane) << 4));
}

static __device__ __forceinline__ float sigm(float x) {
  return 1.0f / (1.0f + __expf(-x));
}

// --- CDNA5 Tensor Data Mover: 1-D contiguous f16 copy, global -> LDS. ------
// Builds the D# in SGPRs (group0: count/lds/global/type=2; group1: data_size=2B,
// tensor_dim0 = tile_dim0 = nelem, stride = nelem) and issues TENSOR_LOAD_TO_LDS.
// Caller must be a single wave; completion via s_wait_tensorcnt (TENSORcnt).
static __device__ __forceinline__ void tdm_load_1d(void* lds, const void* g,
                                                   unsigned nelem /* f16 elems */) {
  unsigned long long ga = (unsigned long long)(uintptr_t)g;
  unsigned la = (unsigned)(uintptr_t)lds;   // flat LDS addr truncates to DS offset
  u32x4 g0;
  g0[0] = 1u;                                            // count=1, user descriptor
  g0[1] = la;                                            // lds_addr (bytes)
  g0[2] = (unsigned)ga;                                  // global_addr[31:0]
  g0[3] = ((unsigned)(ga >> 32) & 0x01FFFFFFu)           // global_addr[56:32]
        | 0x80000000u;                                   // type=2 ("image")
  u32x8 g1;
  g1[0] = 0x00010000u;                                   // data_size=1 -> 2 bytes
  g1[1] = (nelem & 0xFFFFu) << 16;                       // tensor_dim0[15:0]
  g1[2] = (nelem >> 16) | (1u << 16);                    // tensor_dim0[31:16], dim1=1
  g1[3] = (nelem & 0xFFFFu) << 16;                       // tile_dim0 = nelem
  g1[4] = 1u;                                            // tile_dim1=1, tile_dim2=0
  g1[5] = nelem;                                         // tensor_dim0_stride[31:0]
  g1[6] = 0u;
  g1[7] = 0u;
  asm volatile("tensor_load_to_lds %0, %1" :: "s"(g0), "s"(g1) : "memory");
}

// --- CDNA5 async load to LDS: per-lane 16B global->LDS, ASYNCcnt tracked. --
static __device__ __forceinline__ void async_b128(void* lds, const void* g) {
  unsigned la = (unsigned)(uintptr_t)lds;
  unsigned long long ga = (unsigned long long)(uintptr_t)g;
  asm volatile("global_load_async_to_lds_b128 %0, %1, off"
               :: "v"(la), "v"(ga) : "memory");
}
static __device__ __forceinline__ void wait_async0() {
  asm volatile("s_wait_asynccnt 0x0" ::: "memory");
}

static __device__ __forceinline__ void grid_barrier(unsigned* cnt, unsigned& phase) {
  __threadfence();
  __syncthreads();
  if (threadIdx.x == 0) {
    atomicAdd(cnt, 1u);
    phase += NWG;
    volatile unsigned* vc = (volatile unsigned*)cnt;
    while (*vc < phase) { __builtin_amdgcn_s_sleep(8); }
  }
  __syncthreads();
}

// ---------------------------------------------------------------------------
// Prep kernels
// ---------------------------------------------------------------------------
__global__ void init_kernel(unsigned* cnt) {
  if (threadIdx.x == 0 && blockIdx.x == 0) *cnt = 0u;
}

__global__ void bias_kernel(const float* __restrict__ bi_e, const float* __restrict__ bh_e,
                            const float* __restrict__ bi_d, const float* __restrict__ bh_d,
                            float* __restrict__ be, float* __restrict__ bd) {
  int i = blockIdx.x * blockDim.x + threadIdx.x;
  if (i < GATES) { be[i] = bi_e[i] + bh_e[i]; bd[i] = bi_d[i] + bh_d[i]; }
}

// Convert whole signal to f16 once (streamed; trivial vs 23.3 TB/s).
__global__ void cvt_signal_kernel(const float* __restrict__ src,
                                  _Float16* __restrict__ dst, int total) {
  int i = blockIdx.x * blockDim.x + threadIdx.x;
  if (i < total) dst[i] = (_Float16)src[i];
}

// Pack W [N rows, K cols] (row = output column of the GEMM, B = W^T) into
// per-lane WMMA B-fragment order: [wg][kt][ntl][lane][16 halves].
// Gate tile mapping: global n_tile = ntl * nwg_part + wg  (ntl = gate id).
__global__ void pack_b_kernel(const float* __restrict__ W, _Float16* __restrict__ dst,
                              int K, int nwg_part, int total) {
  int idx = blockIdx.x * blockDim.x + threadIdx.x;
  if (idx >= total) return;
  int j    = idx & 15;
  int lane = (idx >> 4) & 31;
  int ntl  = (idx >> 9) & 3;
  int rest = idx >> 11;
  int KT   = K >> 5;
  int kt   = rest % KT;
  int wg   = rest / KT;
  int n = (ntl * nwg_part + wg) * 16 + (lane & 15);
  int k = kt * 32 + ((lane >> 4) << 3) + (j < 8 ? j : j + 8);
  dst[idx] = (_Float16)W[(size_t)n * K + k];
}

// ---------------------------------------------------------------------------
// Persistent seq2seq kernel: 32 WGs x 256 threads, weights resident in LDS
// (staged via TDM), double-buffered f16 hidden state in global (L2-resident),
// x_t staged via async-to-LDS, one grid barrier per encoder step, two per decoder step.
// ---------------------------------------------------------------------------
__global__ void __launch_bounds__(BLK, 1)
seq2seq_kernel(const float* __restrict__ signal,
               const _Float16* __restrict__ sigf16,  // may be null
               const int*   __restrict__ target,
               const float* __restrict__ dec_W_ih,   // [2048,64] gather source
               const float* __restrict__ out_b,      // [64]
               const _Float16* __restrict__ encWih_fr,
               const _Float16* __restrict__ encWhh_fr,
               const _Float16* __restrict__ decWhh_fr,
               const _Float16* __restrict__ outW_fr,
               const float* __restrict__ bias_enc,
               const float* __restrict__ bias_dec,
               _Float16* __restrict__ hbuf,          // 2 * [64,512] f16
               int* __restrict__ tokbuf,             // [64]
               unsigned* __restrict__ cnt,
               float* __restrict__ out)              // [B,T,V]
{
  extern __shared__ char smem[];
  _Float16* whhL  = (_Float16*)(smem + SM_WHH);
  _Float16* auxL  = (_Float16*)(smem + SM_AUX);
  _Float16* xbuf  = (_Float16*)(smem + SM_XLOG);
  float*    lgts  = (float*)   (smem + SM_XLOG);
  float*    gbuf  = (float*)   (smem + SM_GBUF);
  float*    cbuf  = (float*)   (smem + SM_CBUF);
  float*    biasL = (float*)   (smem + SM_BIAS);

  const int wg    = blockIdx.x;
  const int tid   = threadIdx.x;
  const int lane  = tid & 31;
  const int wave  = tid >> 5;
  const int m_tile = wave & 3;           // batch tile 0..3
  const int ntl0  = (wave >> 2) * 2;     // gate tiles {0,1} or {2,3}
  const int ntl1  = ntl0 + 1;
  const int hbase = wg * 16;             // owned hidden columns
  unsigned phase = 0;

  _Float16* hA = hbuf;                   // read buffer
  _Float16* hB = hbuf + B_ * H_;         // write buffer

  // ---- init: zero owned h/c; TDM-stage encoder weight slices into LDS ----
  for (int i = tid; i < B_ * 16; i += BLK) {
    int b = i >> 4, jl = i & 15;
    hA[b * H_ + hbase + jl] = (_Float16)0.f;
    cbuf[i] = 0.f;
  }
  if (wave == 0) {
    tdm_load_1d(whhL, encWhh_fr + (size_t)wg * 32768, 32768u);  // 64KB W_hh slice
    tdm_load_1d(auxL, encWih_fr + (size_t)wg * 8192, 8192u);    // 16KB W_ih slice
    __builtin_amdgcn_s_wait_tensorcnt(0);
  }
  if (tid < 64) biasL[tid] = bias_enc[(tid >> 4) * H_ + hbase + (tid & 15)];
  grid_barrier(cnt, phase);
  __builtin_amdgcn_s_cluster_barrier();  // NOP when not dispatched as a cluster

  // ================= encoder: 1024 recurrent steps =================
  for (int t = 0; t < S_; ++t) {
    if (sigf16) {
      // async per-lane 16B copies of the pre-converted f16 rows
      for (int i = tid; i < (B_ * F_) / 8; i += BLK) {   // 1024 x 16B chunks
        int b = i >> 4, c = i & 15;
        async_b128(xbuf + b * F_ + c * 8,
                   sigf16 + (size_t)(b * S_ + t) * F_ + c * 8);
      }
      wait_async0();
      if (t + 1 < S_ && tid < B_)
        __builtin_prefetch(&sigf16[(size_t)(tid * S_ + t + 1) * F_], 0, 1);
    } else {
      for (int i = tid; i < B_ * F_; i += BLK) {
        int b = i >> 7, f = i & 127;
        xbuf[i] = (_Float16)signal[(size_t)(b * S_ + t) * F_ + f];
        if (t + 1 < S_)
          __builtin_prefetch(&signal[(size_t)(b * S_ + t + 1) * F_ + f], 0, 1);
      }
    }
    __syncthreads();

    v8f a0 = {}, a1 = {};
    for (int kt = 0; kt < (F_ >> 5); ++kt) {        // x @ W_ih^T
      v16h a = load_a_frag(xbuf, F_, m_tile, kt, lane);
      a0 = wmma_f16(a, load_b_frag(auxL, kt, ntl0, lane), a0);
      a1 = wmma_f16(a, load_b_frag(auxL, kt, ntl1, lane), a1);
    }
    for (int kt = 0; kt < (H_ >> 5); ++kt) {        // h @ W_hh^T
      v16h a = load_a_frag(hA, H_, m_tile, kt, lane);
      a0 = wmma_f16(a, load_b_frag(whhL, kt, ntl0, lane), a0);
      a1 = wmma_f16(a, load_b_frag(whhL, kt, ntl1, lane), a1);
    }
    {
      int col = lane & 15, rowoff = (lane >> 4) << 3;
      for (int r = 0; r < 8; ++r) {
        int row = m_tile * 16 + rowoff + r;
        gbuf[(ntl0 * B_ + row) * 16 + col] = a0[r];
        gbuf[(ntl1 * B_ + row) * 16 + col] = a1[r];
      }
    }
    __syncthreads();

    // pointwise LSTM cell for owned 64x16 slice
    for (int e = tid; e < B_ * 16; e += BLK) {
      int b = e >> 4, jl = e & 15;
      float gi = gbuf[(0 * B_ + b) * 16 + jl] + biasL[jl];
      float gf = gbuf[(1 * B_ + b) * 16 + jl] + biasL[16 + jl];
      float gg = gbuf[(2 * B_ + b) * 16 + jl] + biasL[32 + jl];
      float go = gbuf[(3 * B_ + b) * 16 + jl] + biasL[48 + jl];
      float c  = cbuf[e];
      float cn = sigm(gf) * c + sigm(gi) * tanhf(gg);
      float hn = sigm(go) * tanhf(cn);
      cbuf[e] = cn;
      hB[b * H_ + hbase + jl] = (_Float16)hn;
    }
    grid_barrier(cnt, phase);
    _Float16* tmp = hA; hA = hB; hB = tmp;
  }

  // ---- swap LDS to decoder weights (TDM); WG0 stages out_W + seed tokens ----
  if (wave == 0) {
    tdm_load_1d(whhL, decWhh_fr + (size_t)wg * 32768, 32768u);
    if (wg == 0) tdm_load_1d(auxL, outW_fr, 32768u);
    __builtin_amdgcn_s_wait_tensorcnt(0);
  }
  if (wg == 0 && tid < B_) tokbuf[tid] = target[tid * T_];   // teacher start
  if (tid < 64) biasL[tid] = bias_dec[(tid >> 4) * H_ + hbase + (tid & 15)];
  grid_barrier(cnt, phase);

  // ================= decoder: 64 greedy steps =================
  for (int t = 0; t < T_; ++t) {
    v8f a0 = {}, a1 = {};
    for (int kt = 0; kt < (H_ >> 5); ++kt) {        // h @ dec_W_hh^T
      v16h a = load_a_frag(hA, H_, m_tile, kt, lane);
      a0 = wmma_f16(a, load_b_frag(whhL, kt, ntl0, lane), a0);
      a1 = wmma_f16(a, load_b_frag(whhL, kt, ntl1, lane), a1);
    }
    {
      int col = lane & 15, rowoff = (lane >> 4) << 3;
      for (int r = 0; r < 8; ++r) {
        int row = m_tile * 16 + rowoff + r;
        gbuf[(ntl0 * B_ + row) * 16 + col] = a0[r];
        gbuf[(ntl1 * B_ + row) * 16 + col] = a1[r];
      }
    }
    __syncthreads();

    // cell update; one-hot input => gather column of dec_W_ih (relu is no-op)
    for (int e = tid; e < B_ * 16; e += BLK) {
      int b = e >> 4, jl = e & 15;
      int tok = tokbuf[b];
      float gi = gbuf[(0 * B_ + b) * 16 + jl] + biasL[jl]
               + dec_W_ih[(size_t)(0 * H_ + hbase + jl) * V_ + tok];
      float gf = gbuf[(1 * B_ + b) * 16 + jl] + biasL[16 + jl]
               + dec_W_ih[(size_t)(1 * H_ + hbase + jl) * V_ + tok];
      float gg = gbuf[(2 * B_ + b) * 16 + jl] + biasL[32 + jl]
               + dec_W_ih[(size_t)(2 * H_ + hbase + jl) * V_ + tok];
      float go = gbuf[(3 * B_ + b) * 16 + jl] + biasL[48 + jl]
               + dec_W_ih[(size_t)(3 * H_ + hbase + jl) * V_ + tok];
      float c  = cbuf[e];
      float cn = sigm(gf) * c + sigm(gi) * tanhf(gg);
      float hn = sigm(go) * tanhf(cn);
      cbuf[e] = cn;
      hB[b * H_ + hbase + jl] = (_Float16)hn;
    }
    grid_barrier(cnt, phase);   // new h visible device-wide

    // WG0: logits = h @ out_W^T + out_b; argmax -> next token; log_softmax
    if (wg == 0) {
      v8f c0 = {}, c1 = {};
      for (int kt = 0; kt < (H_ >> 5); ++kt) {
        v16h a = load_a_frag(hB, H_, m_tile, kt, lane);
        c0 = wmma_f16(a, load_b_frag(auxL, kt, ntl0, lane), c0);
        c1 = wmma_f16(a, load_b_frag(auxL, kt, ntl1, lane), c1);
      }
      int col = lane & 15, rowoff = (lane >> 4) << 3;
      for (int r = 0; r < 8; ++r) {
        int row = m_tile * 16 + rowoff + r;
        lgts[row * V_ + ntl0 * 16 + col] = c0[r];
        lgts[row * V_ + ntl1 * 16 + col] = c1[r];
      }
      __syncthreads();
      if (tid < B_) {
        int b = tid;
        float mx = -3.4e38f; int am = 0;
        for (int v = 0; v < V_; ++v) {
          float x = lgts[b * V_ + v] + out_b[v];
          if (x > mx) { mx = x; am = v; }
        }
        tokbuf[b] = am;
        float s = 0.f;
        for (int v = 0; v < V_; ++v)
          s += __expf(lgts[b * V_ + v] + out_b[v] - mx);
        float lse = mx + __logf(s);
        for (int v = 0; v < V_; ++v)
          out[((size_t)b * T_ + t) * V_ + v] = lgts[b * V_ + v] + out_b[v] - lse;
      }
      __syncthreads();
    }
    grid_barrier(cnt, phase);   // token visible to all WGs
    _Float16* tmp = hA; hA = hB; hB = tmp;
  }
}

// ---------------------------------------------------------------------------
extern "C" void kernel_launch(void* const* d_in, const int* in_sizes, int n_in,
                              void* d_out, int out_size, void* d_ws, size_t ws_size,
                              hipStream_t stream) {
  (void)in_sizes; (void)n_in; (void)out_size;
  const float* signal   = (const float*)d_in[0];
  const int*   target   = (const int*)  d_in[1];
  const float* enc_W_ih = (const float*)d_in[2];
  const float* enc_W_hh = (const float*)d_in[3];
  const float* enc_b_ih = (const float*)d_in[4];
  const float* enc_b_hh = (const float*)d_in[5];
  const float* dec_W_ih = (const float*)d_in[6];
  const float* dec_W_hh = (const float*)d_in[7];
  const float* dec_b_ih = (const float*)d_in[8];
  const float* dec_b_hh = (const float*)d_in[9];
  const float* out_W    = (const float*)d_in[10];
  const float* out_b    = (const float*)d_in[11];

  char* ws = (char*)d_ws;
  size_t off = 0;
  auto take = [&](size_t bytes) -> char* {
    char* p = ws + off;
    off = (off + bytes + 255) & ~(size_t)255;
    return p;
  };
  unsigned* cnt        = (unsigned*) take(256);
  _Float16* hbuf       = (_Float16*) take((size_t)2 * B_ * H_ * 2);
  int*      tokbuf     = (int*)      take(B_ * sizeof(int));
  _Float16* encWih_fr  = (_Float16*) take((size_t)GATES * F_ * 2);
  _Float16* encWhh_fr  = (_Float16*) take((size_t)GATES * H_ * 2);
  _Float16* decWhh_fr  = (_Float16*) take((size_t)GATES * H_ * 2);
  _Float16* outW_fr    = (_Float16*) take((size_t)V_ * H_ * 2);
  float*    bias_e     = (float*)    take(GATES * sizeof(float));
  float*    bias_d     = (float*)    take(GATES * sizeof(float));

  const size_t sig_bytes = (size_t)B_ * S_ * F_ * 2;
  _Float16* sigf16 = nullptr;
  if (ws_size >= off + sig_bytes + 256)
    sigf16 = (_Float16*)take(sig_bytes);

  init_kernel<<<1, 32, 0, stream>>>(cnt);
  bias_kernel<<<(GATES + 255) / 256, 256, 0, stream>>>(enc_b_ih, enc_b_hh,
                                                       dec_b_ih, dec_b_hh,
                                                       bias_e, bias_d);
  pack_b_kernel<<<(GATES * F_ + 255) / 256, 256, 0, stream>>>(enc_W_ih, encWih_fr,
                                                              F_, NWG, GATES * F_);
  pack_b_kernel<<<(GATES * H_ + 255) / 256, 256, 0, stream>>>(enc_W_hh, encWhh_fr,
                                                              H_, NWG, GATES * H_);
  pack_b_kernel<<<(GATES * H_ + 255) / 256, 256, 0, stream>>>(dec_W_hh, decWhh_fr,
                                                              H_, NWG, GATES * H_);
  pack_b_kernel<<<(V_ * H_ + 255) / 256, 256, 0, stream>>>(out_W, outW_fr,
                                                           H_, 1, V_ * H_);
  if (sigf16) {
    int total = B_ * S_ * F_;
    cvt_signal_kernel<<<(total + 255) / 256, 256, 0, stream>>>(signal, sigf16, total);
  }

  seq2seq_kernel<<<NWG, BLK, SMEM_BYTES, stream>>>(
      signal, sigf16, target, dec_W_ih, out_b,
      encWih_fr, encWhh_fr, decWhh_fr, outW_fr,
      bias_e, bias_d, hbuf, tokbuf, cnt, (float*)d_out);
}
